// MultiHeadAttention_9955734192172
// MI455X (gfx1250) — matrix operand order repaired and checked by reading, not compiled, not run on hardware
//
#include <hip/hip_runtime.h>
#include <hip/hip_bf16.h>

// Problem constants (from reference)
#define DM 1024   // d_model
#define NH 16     // heads
#define HD 64     // d_qkv
#define BB 4      // batch
#define LL 2048   // seq len

typedef __bf16 bf16_t;
typedef __attribute__((ext_vector_type(16))) __bf16 v16bf;
typedef __attribute__((ext_vector_type(8)))  __bf16 v8bf;
typedef __attribute__((ext_vector_type(8)))  float  v8f;
typedef int v4i_t __attribute__((vector_size(16)));

static __device__ __forceinline__ v8f wmma_bf16(v16bf a, v16bf b, v8f c) {
  // (neg_a, A, neg_b, B, c_mod, C, reuse_a, reuse_b)
  return __builtin_amdgcn_wmma_f32_16x16x32_bf16(false, a, false, b, (short)0, c, false, false);
}

// A-fragment (16x32 bf16, lane=row, K chunks at g*16 and g*16+8): two 16B loads
static __device__ __forceinline__ v16bf load_a16(const bf16_t* p0, const bf16_t* p1) {
  v8bf lo = *(const v8bf*)p0;
  v8bf hi = *(const v8bf*)p1;
  return __builtin_shufflevector(lo, hi, 0,1,2,3,4,5,6,7,8,9,10,11,12,13,14,15);
}

// lane <-> lane^16 exchange via DS swizzle (group-of-32, xor_mask=0x10, and=0x1f)
static __device__ __forceinline__ float xor16f(float v) {
  return __builtin_bit_cast(float, __builtin_amdgcn_ds_swizzle(__builtin_bit_cast(int, v), 0x401F));
}

// ---------------------------------------------------------------------------
// Async global->LDS staging (CDNA5 GLOBAL_LOAD_ASYNC_TO_LDS_B128, ASYNCcnt),
// with a synchronous fallback if the builtin isn't available.
// ---------------------------------------------------------------------------
#if defined(__has_builtin)
#if __has_builtin(__builtin_amdgcn_global_load_async_to_lds_b128) && \
    __has_builtin(__builtin_amdgcn_s_wait_asynccnt)
#define USE_ASYNC_LDS 1
#endif
#endif

static __device__ __forceinline__ void copy16_g2l(const bf16_t* g, bf16_t* l) {
#ifdef USE_ASYNC_LDS
  __builtin_amdgcn_global_load_async_to_lds_b128(
      (__attribute__((address_space(1))) v4i_t*)(g),
      (__attribute__((address_space(3))) v4i_t*)(l), 0, 0);
#else
  *(v8bf*)l = *(const v8bf*)g;
#endif
}

static __device__ __forceinline__ void wait_lds_tiles() {
#ifdef USE_ASYNC_LDS
  __builtin_amdgcn_s_wait_asynccnt(0);
#endif
  __syncthreads();
}

// ---------------------------------------------------------------------------
// Conversion kernels
// ---------------------------------------------------------------------------
__global__ void cvt_f32_bf16(const float* __restrict__ in, bf16_t* __restrict__ out, int n) {
  for (int i = blockIdx.x * blockDim.x + threadIdx.x; i < n; i += gridDim.x * blockDim.x)
    out[i] = (__bf16)in[i];
}

// w[N][M][H] (f32) -> wt[N][H][M] (bf16): per-head transposed projection weights
__global__ void cvt_w_transpose(const float* __restrict__ in, bf16_t* __restrict__ out) {
  const int n_total = NH * HD * DM;
  for (int i = blockIdx.x * blockDim.x + threadIdx.x; i < n_total; i += gridDim.x * blockDim.x) {
    int n = i >> 16;            // / (HD*DM)
    int rem = i & 65535;
    int h = rem >> 10;          // / DM
    int m = rem & 1023;
    out[i] = (__bf16)in[((size_t)n * DM + m) * HD + h];
  }
}

// w_o[N][H][M] (f32, == [NH*HD][M]) -> wot[M][NH*HD] (bf16)
__global__ void cvt_wo_transpose(const float* __restrict__ in, bf16_t* __restrict__ out) {
  const int n_total = DM * DM;
  for (int i = blockIdx.x * blockDim.x + threadIdx.x; i < n_total; i += gridDim.x * blockDim.x) {
    int m  = i >> 10;
    int nh = i & 1023;
    out[i] = (__bf16)in[(size_t)nh * DM + m];
  }
}

__global__ void cvt_maskbias(const int* __restrict__ mask, float* __restrict__ bias, int n) {
  for (int i = blockIdx.x * blockDim.x + threadIdx.x; i < n; i += gridDim.x * blockDim.x)
    bias[i] = mask[i] != 0 ? 0.0f : -1e9f;
}

// ---------------------------------------------------------------------------
// QKV projection: per wave, one (16-row tile, head, {q,k,v}) -> 16x64 tile
// Q,K stored [B*NH][L][64]; V stored transposed [B*NH][64][L]
// ---------------------------------------------------------------------------
__global__ __launch_bounds__(256) void qkv_proj_kernel(
    const bf16_t* __restrict__ xb,
    const bf16_t* __restrict__ wtq, const bf16_t* __restrict__ wtk,
    const bf16_t* __restrict__ wtv,
    bf16_t* __restrict__ Qb, bf16_t* __restrict__ Kb, bf16_t* __restrict__ Vtb) {
  const int lane = threadIdx.x & 31;
  const int wid  = threadIdx.x >> 5;
  const int w    = blockIdx.x * 8 + wid;          // 0 .. 512*16*3-1
  const int rowtile = w % 512;
  const int rest    = w / 512;
  const int n       = rest % NH;
  const int which   = rest / NH;                  // 0=q 1=k 2=v
  const int c = lane & 15, g = lane >> 4;
  const int r0 = rowtile * 16;                    // global token row

  const bf16_t* wt = (which == 0) ? wtq : (which == 1) ? wtk : wtv;

  v8f acc[4];
  #pragma unroll
  for (int t = 0; t < 4; ++t) acc[t] = (v8f){0,0,0,0,0,0,0,0};

  const bf16_t* Xrow = xb + (size_t)(r0 + c) * DM;
  #pragma unroll 2
  for (int kb = 0; kb < DM; kb += 32) {
    v16bf a = load_a16(Xrow + kb + g * 16, Xrow + kb + g * 16 + 8);
    #pragma unroll
    for (int ct = 0; ct < 4; ++ct) {
      const bf16_t* Bp = wt + ((size_t)n * HD + ct * 16 + c) * DM + kb + g * 16;
      v16bf bfrag = *(const v16bf*)Bp;
      acc[ct] = wmma_bf16(a, bfrag, acc[ct]);
    }
  }

  const int b  = r0 >> 11;          // /L
  const int l0 = r0 & (LL - 1);
  const size_t bn = (size_t)(b * NH + n);

  if (which < 2) {
    bf16_t* dst = (which == 0) ? Qb : Kb;
    #pragma unroll
    for (int ct = 0; ct < 4; ++ct)
      #pragma unroll
      for (int r = 0; r < 8; ++r)
        dst[(bn * LL + l0 + g * 8 + r) * HD + ct * 16 + c] = (__bf16)acc[ct][r];
  } else {
    #pragma unroll
    for (int ct = 0; ct < 4; ++ct) {
      v8bf pk;
      #pragma unroll
      for (int r = 0; r < 8; ++r) pk[r] = (__bf16)acc[ct][r];
      *(v8bf*)(Vtb + (bn * HD + ct * 16 + c) * LL + l0 + g * 8) = pk;
    }
  }
}

// ---------------------------------------------------------------------------
// Flash attention (transposed form), LDS-staged K/V shared by the workgroup.
// Block = 8 waves, all on one (b, head); each wave owns 32 queries (2 tiles).
// K/V 32-key tiles are double-buffered in LDS via async global->LDS copies.
// S^T = K*Q^T ; online softmax per lane (lane column = query);
// ctx^T = V^T * P^T accumulated in registers. Output ctx [B*L][NH*HD] bf16.
// ---------------------------------------------------------------------------
__global__ __launch_bounds__(256) void attn_kernel(
    const bf16_t* __restrict__ Qb, const bf16_t* __restrict__ Kb,
    const bf16_t* __restrict__ Vtb, const float* __restrict__ maskbias,
    bf16_t* __restrict__ ctxb) {
  __shared__ __align__(16) bf16_t Kls[2][32 * HD];   // [key][h]   4KB x2
  __shared__ __align__(16) bf16_t Vls[2][HD * 32];   // [h][key]   4KB x2

  const int tid  = threadIdx.x;
  const int lane = tid & 31;
  const int wid  = tid >> 5;
  const int bid  = blockIdx.x;        // 0..511
  const int bn   = bid >> 3;          // (b*NH + n)
  const int qb   = bid & 7;           // 256-query block within L
  const int b    = bn >> 4;
  const int c = lane & 15, g = lane >> 4;
  const int q0 = qb * 256 + wid * 32;
  const float scale = 0.125f;         // 1/sqrt(64)

  // Per-thread slice of the cooperative tile copy (16B each for K and V)
  const int kkey = tid >> 3;          // 0..31
  const int kh   = (tid & 7) * 8;     // 0..56
  const int vh   = tid >> 2;          // 0..63
  const int vk   = (tid & 3) * 8;     // 0..24

  const bf16_t* Kgbase = Kb  + (size_t)bn * LL * HD;
  const bf16_t* Vgbase = Vtb + (size_t)bn * HD * LL;

  auto issue_tile = [&](int kb0, int s) {
    copy16_g2l(Kgbase + (size_t)(kb0 + kkey) * HD + kh, &Kls[s][kkey * HD + kh]);
    copy16_g2l(Vgbase + (size_t)vh * LL + kb0 + vk,     &Vls[s][vh * 32 + vk]);
  };

  // Q as B-fragments: lane col = query, K dim = h (two chunks of 32), 2 q-tiles
  v16bf qf[2][2];
  #pragma unroll
  for (int qt = 0; qt < 2; ++qt) {
    const bf16_t* Qrow = Qb + ((size_t)bn * LL + q0 + qt * 16 + c) * HD;
    qf[qt][0] = *(const v16bf*)(Qrow + g * 16);
    qf[qt][1] = *(const v16bf*)(Qrow + 32 + g * 16);
  }

  v8f acc[2][4];
  #pragma unroll
  for (int qt = 0; qt < 2; ++qt)
    #pragma unroll
    for (int t = 0; t < 4; ++t) acc[qt][t] = (v8f){0,0,0,0,0,0,0,0};
  float m[2]    = {-1e30f, -1e30f};
  float lsum[2] = {0.0f, 0.0f};

  const float* mbase = maskbias + (size_t)b * LL;

  issue_tile(0, 0);
  for (int it = 0; it < LL / 32; ++it) {
    const int kb0 = it * 32;
    const int s   = it & 1;
    wait_lds_tiles();                           // tile `it` resident in LDS
    if (it + 1 < LL / 32) issue_tile(kb0 + 32, s ^ 1);  // overlap next DMA

    // ---- scores S^T for 32 keys x 32 queries --------------------------------
    float pv[2][2][8];
    #pragma unroll
    for (int kt = 0; kt < 2; ++kt) {
      const bf16_t* Kp = &Kls[s][(kt * 16 + c) * HD];          // lane row = key
      v16bf a0 = load_a16(Kp + g * 16,      Kp + g * 16 + 8);   // h 0..31
      v16bf a1 = load_a16(Kp + 32 + g * 16, Kp + 40 + g * 16);   // h 32..63
      v8f bias = *(const v8f*)(mbase + kb0 + kt * 16 + g * 8); // key-indexed
      #pragma unroll
      for (int qt = 0; qt < 2; ++qt) {
        v8f sc = (v8f){0,0,0,0,0,0,0,0};
        sc = wmma_bf16(a0, qf[qt][0], sc);
        sc = wmma_bf16(a1, qf[qt][1], sc);
        #pragma unroll
        for (int r = 0; r < 8; ++r) pv[qt][kt][r] = sc[r] * scale + bias[r];
      }
    }

    // ---- online softmax + P^T build per query tile --------------------------
    v16bf pt[2];
    #pragma unroll
    for (int qt = 0; qt < 2; ++qt) {
      float tmax = pv[qt][0][0];
      #pragma unroll
      for (int r = 0; r < 8; ++r)
        tmax = fmaxf(tmax, fmaxf(pv[qt][0][r], pv[qt][1][r]));
      tmax = fmaxf(tmax, xor16f(tmax));
      const float mn = fmaxf(m[qt], tmax);
      const float corr = __expf(m[qt] - mn);
      m[qt] = mn;
      float rs = 0.0f;
      #pragma unroll
      for (int kt = 0; kt < 2; ++kt)
        #pragma unroll
        for (int r = 0; r < 8; ++r) {
          pv[qt][kt][r] = __expf(pv[qt][kt][r] - mn);
          rs += pv[qt][kt][r];
        }
      rs += xor16f(rs);
      lsum[qt] = lsum[qt] * corr + rs;
      #pragma unroll
      for (int t = 0; t < 4; ++t) acc[qt][t] = acc[qt][t] * corr;

      // lanes 0-15 need keys 0-15, lanes 16-31 keys 16-31
      float o0[8], o1[8];
      #pragma unroll
      for (int r = 0; r < 8; ++r) {
        o0[r] = xor16f(pv[qt][0][r]);
        o1[r] = xor16f(pv[qt][1][r]);
      }
      #pragma unroll
      for (int r = 0; r < 8; ++r) {
        pt[qt][r]     = (__bf16)(g ? o1[r]        : pv[qt][0][r]); // K = g*16+r
        pt[qt][8 + r] = (__bf16)(g ? pv[qt][1][r] : o0[r]);        // K = g*16+8+r
      }
    }

    // ---- ctx^T += V^T tile x P^T (4 h-tiles, V-frag shared by both q-tiles) --
    #pragma unroll
    for (int t = 0; t < 4; ++t) {
      const bf16_t* Vp = &Vls[s][(t * 16 + c) * 32];             // lane row = h
      v16bf vf = load_a16(Vp + g * 16, Vp + g * 16 + 8);
      #pragma unroll
      for (int qt = 0; qt < 2; ++qt)
        acc[qt][t] = wmma_bf16(vf, pt[qt], acc[qt][t]);
    }
  }

  // normalize and store ctx row-major [B*L][NH*HD] (lane col = query row)
  #pragma unroll
  for (int qt = 0; qt < 2; ++qt) {
    const float inv = 1.0f / lsum[qt];
    bf16_t* crow = ctxb + ((size_t)b * LL + q0 + qt * 16 + c) * DM + (bn & 15) * HD;
    #pragma unroll
    for (int t = 0; t < 4; ++t) {
      v8bf pk;
      #pragma unroll
      for (int r = 0; r < 8; ++r) pk[r] = (__bf16)(acc[qt][t][r] * inv);
      *(v8bf*)(crow + t * 16 + g * 8) = pk;
    }
  }
}

// ---------------------------------------------------------------------------
// Output projection: out[B*L][DM] f32 = ctx[B*L][DM] x Wo[DM][DM]
// One wave = 16 rows x 64 cols.
// ---------------------------------------------------------------------------
__global__ __launch_bounds__(256) void out_proj_kernel(
    const bf16_t* __restrict__ ctxb, const bf16_t* __restrict__ wot,
    float* __restrict__ out) {
  const int lane = threadIdx.x & 31;
  const int wid  = threadIdx.x >> 5;
  const int w    = blockIdx.x * 8 + wid;          // 0 .. 8191
  const int rowtile = w & 511;
  const int ctile   = w >> 9;                     // 0..15 -> col base ctile*64
  const int c = lane & 15, g = lane >> 4;
  const int r0 = rowtile * 16;

  v8f acc[4];
  #pragma unroll
  for (int t = 0; t < 4; ++t) acc[t] = (v8f){0,0,0,0,0,0,0,0};

  const bf16_t* Xrow = ctxb + (size_t)(r0 + c) * DM;
  #pragma unroll 2
  for (int kb = 0; kb < DM; kb += 32) {
    v16bf a = load_a16(Xrow + kb + g * 16, Xrow + kb + g * 16 + 8);
    #pragma unroll
    for (int ct = 0; ct < 4; ++ct) {
      const bf16_t* Bp = wot + ((size_t)(ctile * 64 + ct * 16 + c)) * DM + kb + g * 16;
      v16bf bfrag = *(const v16bf*)Bp;
      acc[ct] = wmma_bf16(a, bfrag, acc[ct]);
    }
  }

  #pragma unroll
  for (int ct = 0; ct < 4; ++ct)
    #pragma unroll
    for (int r = 0; r < 8; ++r)
      out[(size_t)(r0 + g * 8 + r) * DM + ctile * 64 + ct * 16 + c] = acc[ct][r];
}

// ---------------------------------------------------------------------------
extern "C" void kernel_launch(void* const* d_in, const int* in_sizes, int n_in,
                              void* d_out, int out_size, void* d_ws, size_t ws_size,
                              hipStream_t stream) {
  (void)in_sizes; (void)n_in; (void)out_size; (void)ws_size;
  const float* x    = (const float*)d_in[0];
  const int*   mask = (const int*)d_in[1];
  const float* w_q  = (const float*)d_in[2];
  const float* w_k  = (const float*)d_in[3];
  const float* w_v  = (const float*)d_in[4];
  const float* w_o  = (const float*)d_in[5];
  float* out = (float*)d_out;

  char* ws = (char*)d_ws;
  size_t off = 0;
  auto alloc = [&](size_t bytes) -> char* {
    char* p = ws + off;
    off += (bytes + 255) & ~(size_t)255;
    return p;
  };
  bf16_t* xb   = (bf16_t*)alloc((size_t)BB * LL * DM * 2);       // 16 MB
  bf16_t* Qb   = (bf16_t*)alloc((size_t)BB * NH * LL * HD * 2);  // 16 MB
  bf16_t* Kb   = (bf16_t*)alloc((size_t)BB * NH * LL * HD * 2);  // 16 MB
  bf16_t* Vtb  = (bf16_t*)alloc((size_t)BB * NH * HD * LL * 2);  // 16 MB
  bf16_t* ctxb = (bf16_t*)alloc((size_t)BB * LL * DM * 2);       // 16 MB
  bf16_t* wtq  = (bf16_t*)alloc((size_t)NH * HD * DM * 2);       // 2 MB
  bf16_t* wtk  = (bf16_t*)alloc((size_t)NH * HD * DM * 2);
  bf16_t* wtv  = (bf16_t*)alloc((size_t)NH * HD * DM * 2);
  bf16_t* wot  = (bf16_t*)alloc((size_t)DM * DM * 2);
  float*  mb   = (float*)alloc((size_t)BB * LL * 4);

  // 1) conversions
  cvt_f32_bf16<<<4096, 256, 0, stream>>>(x, xb, BB * LL * DM);
  cvt_w_transpose<<<1024, 256, 0, stream>>>(w_q, wtq);
  cvt_w_transpose<<<1024, 256, 0, stream>>>(w_k, wtk);
  cvt_w_transpose<<<1024, 256, 0, stream>>>(w_v, wtv);
  cvt_wo_transpose<<<1024, 256, 0, stream>>>(w_o, wot);
  cvt_maskbias<<<32, 256, 0, stream>>>(mask, mb, BB * LL);

  // 2) QKV projections: 512 rowtiles * 16 heads * 3 = 24576 waves / 8 per block
  qkv_proj_kernel<<<3072, 256, 0, stream>>>(xb, wtq, wtk, wtv, Qb, Kb, Vtb);

  // 3) flash attention: 64 (b,head) x 8 query blocks = 512 workgroups
  attn_kernel<<<512, 256, 0, stream>>>(Qb, Kb, Vtb, mb, ctxb);

  // 4) output projection: 8192 waves / 8 per block
  out_proj_kernel<<<1024, 256, 0, stream>>>(ctxb, wot, out);
}
